// MultiHead_12902081757531
// MI455X (gfx1250) — matrix operand order repaired and checked by reading, not compiled, not run on hardware
//
#include <hip/hip_runtime.h>

#define EMB   768
#define NHEAD 12
#define BATCH 32
#define SEQ   1024
#define HDIM  64

typedef __attribute__((ext_vector_type(16))) __bf16 v16bf;
typedef __attribute__((ext_vector_type(8)))  __bf16 v8bf;
typedef __attribute__((ext_vector_type(8)))  float  v8f;

__device__ __forceinline__ unsigned short f2bf(float f) {
  return __builtin_bit_cast(unsigned short, (__bf16)f);   // native cvt on gfx1250
}
__device__ __forceinline__ unsigned pack_bf2(float lo, float hi) {
  return (unsigned)f2bf(lo) | ((unsigned)f2bf(hi) << 16);
}
__device__ __forceinline__ v16bf cat16(v8bf lo, v8bf hi) {
  return __builtin_shufflevector(lo, hi, 0,1,2,3,4,5,6,7,8,9,10,11,12,13,14,15);
}

// A operand (16x32), LDS tile [M][K] row-major, p = row base (lane-resolved).
// lane group g: t=0..7 -> K=8g+t ; t=8..15 -> K=16+8g+(t-8): two b128 loads.
__device__ __forceinline__ v16bf lds_a32(const unsigned short* p) {
  const int g = (threadIdx.x >> 4) & 1;
  v8bf lo = *(const v8bf*)(p + 8 * g);
  v8bf hi = *(const v8bf*)(p + 16 + 8 * g);
  return cat16(lo, hi);
}

// B operand (32x16), LDS tile [N][K] row-major, p = N-row base (lane-resolved).
// lane group g: t=0..7 -> K=8g+t ; t=8..15 -> K=16+8g+(t-8): two b128 loads.
__device__ __forceinline__ v16bf lds_b32(const unsigned short* p) {
  const int g = (threadIdx.x >> 4) & 1;
  v8bf lo = *(const v8bf*)(p + 8 * g);
  v8bf hi = *(const v8bf*)(p + 16 + 8 * g);
  return cat16(lo, hi);
}

__device__ __forceinline__ v8f wmma_bf16(v16bf a, v16bf b, v8f c) {
  return __builtin_amdgcn_wmma_f32_16x16x32_bf16(false, a, false, b, (short)0, c, false, false);
}

// ---------------------------------------------------------------------------
// Pre-pass A: x fp32 -> bf16, once (avoids 36x re-conversion in the GEMM).
// grid = B*N*EMB / (256*8) blocks.
// ---------------------------------------------------------------------------
__global__ __launch_bounds__(256) void xcvt_kernel(
    const float* __restrict__ x, unsigned short* __restrict__ Xb)
{
  size_t base = ((size_t)blockIdx.x * 256 + threadIdx.x) * 8;
  float4 a = *(const float4*)(x + base);
  float4 b = *(const float4*)(x + base + 4);
  uint4 o;
  o.x = pack_bf2(a.x, a.y);
  o.y = pack_bf2(a.z, a.w);
  o.z = pack_bf2(b.x, b.y);
  o.w = pack_bf2(b.z, b.w);
  *(uint4*)(Xb + base) = o;
}

// ---------------------------------------------------------------------------
// Pre-pass B: Wq/Wk/Wv fp32 [k][c] -> bf16 transposed [z][c][k] (B-operand layout).
// ---------------------------------------------------------------------------
__global__ __launch_bounds__(256) void wcvt_kernel(
    const float* __restrict__ Wq, const float* __restrict__ Wk,
    const float* __restrict__ Wv, unsigned short* __restrict__ Wt)
{
  int idx = blockIdx.x * 256 + threadIdx.x;
  int z = idx / (EMB * EMB);
  int rem = idx - z * (EMB * EMB);
  int k = rem / EMB, c = rem - k * EMB;
  const float* W = (z == 0) ? Wq : (z == 1) ? Wk : Wv;
  Wt[(size_t)z * EMB * EMB + (size_t)c * EMB + k] = f2bf(W[rem]);
}

// ---------------------------------------------------------------------------
// Kernel 1: QKV projection from pre-converted bf16 operands.
// grid = (EMB/64, (B*N)/64, 3), block = 256. Inner loop: pure uint4 copies + WMMA.
// z=0: Q -> [b,h,i,e]; z=1: K -> [b,h,j,e]; z=2: V -> [b,h,e,j] (transposed).
// ---------------------------------------------------------------------------
__global__ __launch_bounds__(256) void qkv_proj_kernel(
    const unsigned short* __restrict__ Xb, const unsigned short* __restrict__ Wt,
    const float* __restrict__ bq, const float* __restrict__ bk, const float* __restrict__ bv,
    unsigned short* __restrict__ Qh, unsigned short* __restrict__ Kh, unsigned short* __restrict__ Vt)
{
  __shared__ __align__(16) unsigned short sx[64 * 40];  // [i 0..63][k 0..31]
  __shared__ __align__(16) unsigned short sw[64 * 40];  // [n 0..63][k 0..31]

  const int tid  = threadIdx.x;
  const int lane = tid & 31, wvid = tid >> 5;
  const int rs = wvid >> 1, ch = wvid & 1;
  const int z  = blockIdx.z;
  const int R0 = blockIdx.y * 64;
  const int C0 = blockIdx.x * 64;
  const int n = lane & 15, g = lane >> 4;

  const float* bias = (z == 0) ? bq : (z == 1) ? bk : bv;
  const unsigned short* Wz = Wt + (size_t)z * EMB * EMB;

  v8f acc0 = {0,0,0,0,0,0,0,0};
  v8f acc1 = {0,0,0,0,0,0,0,0};

  const int r = tid >> 2, part = tid & 3;   // 256 uint4 chunks per tile
  for (int kk = 0; kk < EMB; kk += 32) {
    *(uint4*)(sx + r * 40 + part * 8) =
        *(const uint4*)(Xb + (size_t)(R0 + r) * EMB + kk + part * 8);
    *(uint4*)(sw + r * 40 + part * 8) =
        *(const uint4*)(Wz + (size_t)(C0 + r) * EMB + kk + part * 8);
    __syncthreads();
    v16bf a = lds_a32(sx + (16 * rs + n) * 40);
    acc0 = wmma_bf16(a, lds_b32(sw + (32 * ch + 0  + n) * 40), acc0);
    acc1 = wmma_bf16(a, lds_b32(sw + (32 * ch + 16 + n) * 40), acc1);
    __syncthreads();
  }

#pragma unroll
  for (int tt = 0; tt < 2; ++tt) {
    v8f acc = tt ? acc1 : acc0;
    int gcol = C0 + 32 * ch + 16 * tt + n;
    float bval = bias[gcol];
    int h = gcol >> 6, e = gcol & 63;
#pragma unroll
    for (int t = 0; t < 8; ++t) {
      int grow = R0 + 16 * rs + t + 8 * g;
      int b = grow >> 10, nn = grow & 1023;
      unsigned short hv = f2bf(acc[t] + bval);
      size_t bh = (size_t)b * NHEAD + h;
      if (z == 0)      Qh[(bh * SEQ + nn) * HDIM + e] = hv;
      else if (z == 1) Kh[(bh * SEQ + nn) * HDIM + e] = hv;
      else             Vt[(bh * HDIM + e) * SEQ + nn] = hv;
    }
  }
}

// ---------------------------------------------------------------------------
// Kernel 2: per-key-column softmax stats (softmax over query axis i).
// grid = (SEQ/64, H, B), block = 128; wave w owns j-tile w.
// ---------------------------------------------------------------------------
__global__ __launch_bounds__(128) void col_stats_kernel(
    const unsigned short* __restrict__ Qh, const unsigned short* __restrict__ Kh,
    float* __restrict__ Ms, float* __restrict__ Zs)
{
  __shared__ __align__(16) unsigned short sk[64 * 72];  // [j 0..63][e 0..63], resident
  __shared__ __align__(16) unsigned short sq[16 * 72];  // [i 0..15][e 0..63]

  const int tid  = threadIdx.x;
  const int lane = tid & 31, wvid = tid >> 5;
  const int j0 = blockIdx.x * 64;
  const size_t bh = (size_t)blockIdx.z * NHEAD + blockIdx.y;
  const float scale = 0.28867513459481287f;   // 1/sqrt(NHEAD)
  const float NEG_INF = -__builtin_inff();
  const int n = lane & 15, g = lane >> 4;

#pragma unroll
  for (int q = 0; q < 4; ++q) {
    int cix = tid + 128 * q;                 // 512 uint4 chunks
    int j = cix >> 3, part = cix & 7;
    *(uint4*)(sk + j * 72 + part * 8) =
        *(const uint4*)(Kh + (bh * SEQ + j0 + j) * HDIM + part * 8);
  }
  __syncthreads();

  float m = NEG_INF, zacc = 0.0f;
  const unsigned short* arow = sq + n * 72;
  const unsigned short* brow = sk + (16 * wvid + n) * 72;
  const int rr = tid >> 3, rpart = tid & 7;

  for (int i0 = 0; i0 < SEQ; i0 += 16) {
    *(uint4*)(sq + rr * 72 + rpart * 8) =
        *(const uint4*)(Qh + (bh * SEQ + i0 + rr) * HDIM + rpart * 8);
    __syncthreads();
    v8f c = {0,0,0,0,0,0,0,0};
    c = wmma_bf16(lds_a32(arow),      lds_b32(brow),      c);   // e = 0..31
    c = wmma_bf16(lds_a32(arow + 32), lds_b32(brow + 32), c);   // e = 32..63

    float tmax = NEG_INF;
#pragma unroll
    for (int t = 0; t < 8; ++t) tmax = fmaxf(tmax, c[t] * scale);
    float mnew = fmaxf(m, tmax);
    float s = 0.0f;
#pragma unroll
    for (int t = 0; t < 8; ++t) s += __expf(c[t] * scale - mnew);
    zacc = zacc * __expf(m - mnew) + s;
    m = mnew;
    __syncthreads();
  }

  // lanes L and L^16 hold row-groups M=0..7 / 8..15 of the same column j
  float mo = __shfl_xor(m, 16, 32);
  float zo = __shfl_xor(zacc, 16, 32);
  float mc = fmaxf(m, mo);
  float zc = zacc * __expf(m - mc) + zo * __expf(mo - mc);
  if (g == 0) {
    int j = j0 + 16 * wvid + n;
    Ms[bh * SEQ + j] = mc;
    Zs[bh * SEQ + j] = zc;
  }
}

// ---------------------------------------------------------------------------
// Kernel 3: out[i,e] = sum_j exp(scale*S[i,j]-m_j)/Z_j * V[j,e].
// grid = (SEQ/64, H, B), block = 128; wave w owns i-rows [i0+16w, +16).
// ---------------------------------------------------------------------------
__global__ __launch_bounds__(128) void attn_out_kernel(
    const unsigned short* __restrict__ Qh, const unsigned short* __restrict__ Kh,
    const unsigned short* __restrict__ Vt,
    const float* __restrict__ Ms, const float* __restrict__ Zs,
    float* __restrict__ out)
{
  __shared__ __align__(16) unsigned short sq[64 * 72];     // Q block [i][e], once
  __shared__ __align__(16) unsigned short sk[32 * 72];     // K tile [j][e]
  __shared__ __align__(16) unsigned short sv[64 * 40];     // V^T tile [e][j]
  __shared__ __align__(16) unsigned short sp[4 * 16 * 32]; // per-wave P [i][j]
  __shared__ float sm[32];
  __shared__ float sz[32];

  const int tid  = threadIdx.x;
  const int lane = tid & 31, wvid = tid >> 5;
  const int i0 = blockIdx.x * 64;
  const int h = blockIdx.y, b = blockIdx.z;
  const size_t bh = (size_t)b * NHEAD + h;
  const float scale = 0.28867513459481287f;
  const int n = lane & 15, g = lane >> 4;

#pragma unroll
  for (int q = 0; q < 4; ++q) {
    int cix = tid + 128 * q;                 // 512 uint4 chunks
    int rq = cix >> 3, part = cix & 7;
    *(uint4*)(sq + rq * 72 + part * 8) =
        *(const uint4*)(Qh + (bh * SEQ + i0 + rq) * HDIM + part * 8);
  }
  __syncthreads();
  const unsigned short* aqrow = sq + (16 * wvid + n) * 72;
  v16bf aq0 = lds_a32(aqrow);
  v16bf aq1 = lds_a32(aqrow + 32);

  v8f acc[4];
#pragma unroll
  for (int et = 0; et < 4; ++et) acc[et] = (v8f){0,0,0,0,0,0,0,0};

  unsigned short* myp = sp + wvid * (16 * 32);

  for (int j0 = 0; j0 < SEQ; j0 += 32) {
#pragma unroll
    for (int q = 0; q < 2; ++q) {
      int cix = tid + 128 * q;               // 256 uint4 chunks each
      {
        int j = cix >> 3, part = cix & 7;    // K tile [j][e]
        *(uint4*)(sk + j * 72 + part * 8) =
            *(const uint4*)(Kh + (bh * SEQ + j0 + j) * HDIM + part * 8);
      }
      {
        int e = cix >> 2, part = cix & 3;    // V^T tile [e][j]
        *(uint4*)(sv + e * 40 + part * 8) =
            *(const uint4*)(Vt + (bh * HDIM + e) * SEQ + j0 + part * 8);
      }
    }
    if (tid < 32) {
      sm[tid] = Ms[bh * SEQ + j0 + tid];
      sz[tid] = Zs[bh * SEQ + j0 + tid];
    }
    // prefetch next j-step's K/V rows (global_prefetch_b8)
    {
      int jp = (j0 + 32 < SEQ) ? j0 + 32 : j0;
      __builtin_prefetch(Kh + (bh * SEQ + jp + (tid >> 2)) * HDIM, 0, 0);
      __builtin_prefetch(Vt + (bh * HDIM + (tid >> 1)) * SEQ + jp, 0, 0);
    }
    __syncthreads();

    const unsigned short* b0 = sk + n * 72;         // j sub-tile 0, col n
    const unsigned short* b1 = sk + (16 + n) * 72;  // j sub-tile 1
    v8f c0 = {0,0,0,0,0,0,0,0};
    v8f c1 = {0,0,0,0,0,0,0,0};
    c0 = wmma_bf16(aq0, lds_b32(b0),      c0);
    c0 = wmma_bf16(aq1, lds_b32(b0 + 32), c0);
    c1 = wmma_bf16(aq0, lds_b32(b1),      c1);
    c1 = wmma_bf16(aq1, lds_b32(b1 + 32), c1);

    // P = exp(scale*S - m_j)/Z_j; relayout D-format -> A-format via wave-private LDS
#pragma unroll
    for (int sub = 0; sub < 2; ++sub) {
      v8f c = sub ? c1 : c0;
      int jl = sub * 16 + n;
      float mj = sm[jl];
      float rz = 1.0f / sz[jl];
#pragma unroll
      for (int t = 0; t < 8; ++t) {
        float pv = __expf(c[t] * scale - mj) * rz;
        myp[(t + 8 * g) * 32 + jl] = f2bf(pv);
      }
    }
    v16bf ap = lds_a32(myp + n * 32);
#pragma unroll
    for (int et = 0; et < 4; ++et)
      acc[et] = wmma_bf16(ap, lds_b32(sv + (16 * et + n) * 40), acc[et]);
    __syncthreads();
  }

#pragma unroll
  for (int et = 0; et < 4; ++et) {
    int e = 16 * et + n;
#pragma unroll
    for (int t = 0; t < 8; ++t) {
      int i = i0 + 16 * wvid + t + 8 * g;
      out[((size_t)b * SEQ + i) * EMB + h * HDIM + e] = acc[et][t];
    }
  }
}

extern "C" void kernel_launch(void* const* d_in, const int* in_sizes, int n_in,
                              void* d_out, int out_size, void* d_ws, size_t ws_size,
                              hipStream_t stream) {
  const float* x  = (const float*)d_in[0];
  const float* Wq = (const float*)d_in[1];
  const float* bq = (const float*)d_in[2];
  const float* Wk = (const float*)d_in[3];
  const float* bk = (const float*)d_in[4];
  const float* Wv = (const float*)d_in[5];
  const float* bv = (const float*)d_in[6];
  float* out = (float*)d_out;

  // Workspace: Q,K,V^T bf16 (3x48MB) + Xb bf16 (48MB) + Wt bf16 (3.4MB) + stats (3MB)
  const size_t QSZ  = (size_t)BATCH * NHEAD * SEQ * HDIM;   // 25,165,824
  const size_t XSZ  = (size_t)BATCH * SEQ * EMB;            // 25,165,824
  const size_t WTSZ = (size_t)3 * EMB * EMB;                // 1,769,472
  unsigned short* Qh = (unsigned short*)d_ws;
  unsigned short* Kh = Qh + QSZ;
  unsigned short* Vt = Kh + QSZ;
  unsigned short* Xb = Vt + QSZ;
  unsigned short* Wt = Xb + XSZ;
  float* Ms = (float*)(Wt + WTSZ);
  float* Zs = Ms + (size_t)BATCH * NHEAD * SEQ;

  xcvt_kernel<<<(unsigned)(XSZ / (256 * 8)), 256, 0, stream>>>(x, Xb);
  wcvt_kernel<<<(unsigned)(WTSZ / 256), 256, 0, stream>>>(Wq, Wk, Wv, Wt);
  qkv_proj_kernel<<<dim3(EMB / 64, (BATCH * SEQ) / 64, 3), 256, 0, stream>>>(
      Xb, Wt, bq, bk, bv, Qh, Kh, Vt);
  col_stats_kernel<<<dim3(SEQ / 64, NHEAD, BATCH), 128, 0, stream>>>(Qh, Kh, Ms, Zs);
  attn_out_kernel<<<dim3(SEQ / 64, NHEAD, BATCH), 128, 0, stream>>>(Qh, Kh, Vt, Ms, Zs, out);
}
